// GraphAttentionLayer_28174985462253
// MI455X (gfx1250) — matrix operand order repaired and checked by reading, not compiled, not run on hardware
//
#include <hip/hip_runtime.h>
#include <hip/hip_bf16.h>
#include <stdint.h>

#define BB   8
#define NN   2048
#define FIN  256
#define FOUT 128
#define ALPHA 0.2f
#define NEG_INF_VAL -9000000000000000.0f

typedef __attribute__((ext_vector_type(16))) __bf16 v16bf;
typedef __attribute__((ext_vector_type(8)))  float  v8f;

union BF16x16 { v16bf v; uint32_t u[8]; uint4 q[2]; };
union F32x8   { v8f v; float f[8]; };

__device__ __forceinline__ uint16_t f32_to_bf16(float x) {
  union { float f; uint32_t u; } c; c.f = x;
  uint32_t r = c.u + 0x7FFFu + ((c.u >> 16) & 1u);   // round-to-nearest-even
  return (uint16_t)(r >> 16);
}
__device__ __forceinline__ uint32_t pack_bf16(float lo, float hi) {
  return (uint32_t)f32_to_bf16(lo) | ((uint32_t)f32_to_bf16(hi) << 16);
}

// Async global->LDS copy of 16 bytes per lane (CDNA5, tracked by ASYNCcnt).
// lds_off: 32-bit LDS byte address (low 32 bits of the generic shared ptr,
// per the ISA aperture mapping). gaddr: 64-bit global address.
__device__ __forceinline__ void async_copy_b128(void* lds_ptr, const void* gptr) {
  uint32_t lds_off = (uint32_t)(uintptr_t)lds_ptr;
  uint64_t gaddr   = (uint64_t)(uintptr_t)gptr;
  asm volatile("global_load_async_to_lds_b128 %0, %1, off"
               :: "v"(lds_off), "v"(gaddr) : "memory");
}
__device__ __forceinline__ void async_wait0() {
  asm volatile("s_wait_asynccnt 0x0" ::: "memory");
}

// ---------------------------------------------------------------------------
// Kernel 1: Wh[b,n,o] = sum_k h[b,n,k] * W[o,k] + Wb[o]
// bf16 WMMA 16x16x32. Block: 256 thr (8 waves) -> 64 rows x 128 cols tile.
// Outputs: wh (f32, row-major [b][n][o]) and whT (bf16, [b][o][n], n-contig).
// ---------------------------------------------------------------------------
#define HS_STRIDE 264   // 256 + 8 (ushort units), keeps 16B alignment
__global__ __launch_bounds__(256) void gat_wh(
    const float* __restrict__ h, const float* __restrict__ Ww,
    const float* __restrict__ Wb,
    float* __restrict__ wh, uint16_t* __restrict__ whT)
{
  __shared__ __align__(16) uint16_t hS[64  * HS_STRIDE];   //  33.8 KB
  __shared__ __align__(16) uint16_t wS[FOUT * HS_STRIDE];  //  67.6 KB

  const int b     = blockIdx.y;
  const int nbase = blockIdx.x * 64;
  const int tid   = threadIdx.x;
  const int lane  = tid & 31;
  const int wave  = tid >> 5;

  // Stage h tile (64x256) and W (128x256) to LDS as bf16.
  {
    const float* src = h + (size_t)(b * NN + nbase) * FIN;
    for (int q = tid; q < (64 * FIN) / 4; q += 256) {
      int r = q >> 6, c4 = (q & 63) * 4;
      float4 v = *(const float4*)(src + (size_t)r * FIN + c4);
      uint32_t* dst = (uint32_t*)&hS[r * HS_STRIDE + c4];
      dst[0] = pack_bf16(v.x, v.y);
      dst[1] = pack_bf16(v.z, v.w);
    }
    for (int q = tid; q < (FOUT * FIN) / 4; q += 256) {
      int r = q >> 6, c4 = (q & 63) * 4;
      float4 v = *(const float4*)(Ww + (size_t)r * FIN + c4);
      uint32_t* dst = (uint32_t*)&wS[r * HS_STRIDE + c4];
      dst[0] = pack_bf16(v.x, v.y);
      dst[1] = pack_bf16(v.z, v.w);
    }
  }
  __syncthreads();

  const int rowtile = wave & 3;      // 4 row tiles of 16
  const int ogrp    = wave >> 2;     // 2 column groups of 64
  const int lh      = lane >> 4;     // lane half (ISA A/B/C layouts)
  const int l16     = lane & 15;

  F32x8 acc[4];
  for (int ct = 0; ct < 4; ++ct) {
    float bias = Wb[ogrp * 64 + ct * 16 + l16];   // col of C = l16 for all v
    for (int v = 0; v < 8; ++v) acc[ct].f[v] = bias;
  }

  const uint16_t* aBase = &hS[(rowtile * 16 + l16) * HS_STRIDE];
  for (int kk = 0; kk < FIN; kk += 32) {
    // A frag (16x32 bf16): lane<16 -> K kk+{0..7,16..23}; lane>=16 -> +8 / +24
    BF16x16 af;
    af.q[0] = *(const uint4*)(aBase + kk + (lh ? 8 : 0));
    af.q[1] = *(const uint4*)(aBase + kk + (lh ? 24 : 16));
    for (int ct = 0; ct < 4; ++ct) {
      int o = ogrp * 64 + ct * 16 + l16;   // B column = lane (per half)
      const uint16_t* bBase = &wS[o * HS_STRIDE + kk + (lh ? 8 : 0)];
      BF16x16 bf;
      bf.q[0] = *(const uint4*)(bBase);
      bf.q[1] = *(const uint4*)(bBase + 16);
      acc[ct].v = __builtin_amdgcn_wmma_f32_16x16x32_bf16(
          false, af.v, false, bf.v, (short)0, acc[ct].v, false, false);
    }
  }

  for (int ct = 0; ct < 4; ++ct) {
    int o = ogrp * 64 + ct * 16 + l16;
    for (int v = 0; v < 8; ++v) {
      int gn = nbase + rowtile * 16 + (lh ? v + 8 : v);  // C/D row = v / v+8
      float val = acc[ct].f[v];
      wh [(size_t)(b * NN + gn) * FOUT + o] = val;
      whT[(size_t)(b * FOUT + o) * NN + gn] = f32_to_bf16(val);
    }
  }
}

// ---------------------------------------------------------------------------
// Kernel 2: s1[r] = Wh[r,:]·a1 ; s2[r] = Wh[r,:]·a2   (r = b*N+n, 16384 rows)
// ---------------------------------------------------------------------------
__global__ __launch_bounds__(256) void gat_s12(
    const float* __restrict__ wh, const float* __restrict__ aw,
    float* __restrict__ s1, float* __restrict__ s2)
{
  int r = blockIdx.x * 256 + threadIdx.x;
  const float4* row = (const float4*)(wh + (size_t)r * FOUT);
  const float4* a1  = (const float4*)(aw);
  const float4* a2  = (const float4*)(aw + FOUT);
  float acc1 = 0.f, acc2 = 0.f;
  for (int q = 0; q < FOUT / 4; ++q) {
    float4 w = row[q], x = a1[q], y = a2[q];
    acc1 += w.x * x.x + w.y * x.y + w.z * x.z + w.w * x.w;
    acc2 += w.x * y.x + w.y * y.y + w.z * y.z + w.w * y.w;
  }
  s1[r] = acc1;
  s2[r] = acc2;
}

// ---------------------------------------------------------------------------
// Kernel 3: masked leaky-relu scores -> softmax -> attention@Wh -> ELU.
// One block = (b, 16 rows). 8 waves: 2 rows each for softmax, then split
// K(j)=2048 8 ways for the WMMA GEMM, cross-wave reduce through LDS.
// ---------------------------------------------------------------------------
#define EB_STRIDE 2080   // 2048 + 32 floats; 16B-aligned rows
__global__ __launch_bounds__(256) void gat_attn(
    const int* __restrict__ adj, const float* __restrict__ s1,
    const float* __restrict__ s2, const float* __restrict__ ab,
    const uint16_t* __restrict__ whT, float* __restrict__ out)
{
  __shared__ __align__(16) float eBuf[16 * EB_STRIDE];   // 130 KB (CDNA5 LDS)
  __shared__ __align__(16) float s2S[NN];                //   8 KB
  __shared__ float rowSum[16];

  const int b    = blockIdx.y;
  const int i0   = blockIdx.x * 16;
  const int tid  = threadIdx.x;
  const int lane = tid & 31;
  const int wave = tid >> 5;
  const float abv = ab[0];

  // Async-stage the shared s2 row (2048 f32) into LDS: 2 x b128 per thread.
  async_copy_b128(&s2S[tid * 4],        s2 + b * NN + tid * 4);
  async_copy_b128(&s2S[1024 + tid * 4], s2 + b * NN + 1024 + tid * 4);
  async_wait0();
  __syncthreads();

  // --- softmax prologue: each wave owns rows 2w, 2w+1 ---
  for (int rr = 0; rr < 2; ++rr) {
    int r = wave * 2 + rr;
    int i = i0 + r;
    float s1v = s1[b * NN + i] + abv;
    const int4* adjRow = (const int4*)(adj + (size_t)(b * NN + i) * NN);
    float rmax = NEG_INF_VAL;
    for (int j0 = 0; j0 < NN; j0 += 128) {          // 128 cols per wave-iter
      int jb = (j0 >> 2) + lane;                    // int4 index
      __builtin_prefetch((const char*)(adjRow + jb) + 4096, 0, 0);
      int4   a4  = adjRow[jb];
      float4 sv  = *(const float4*)&s2S[j0 + lane * 4];
      float4 e4;
      e4.x = s1v + sv.x; e4.x = fmaxf(e4.x, ALPHA * e4.x);
      e4.y = s1v + sv.y; e4.y = fmaxf(e4.y, ALPHA * e4.y);
      e4.z = s1v + sv.z; e4.z = fmaxf(e4.z, ALPHA * e4.z);
      e4.w = s1v + sv.w; e4.w = fmaxf(e4.w, ALPHA * e4.w);
      e4.x = (a4.x > 0) ? e4.x : NEG_INF_VAL;
      e4.y = (a4.y > 0) ? e4.y : NEG_INF_VAL;
      e4.z = (a4.z > 0) ? e4.z : NEG_INF_VAL;
      e4.w = (a4.w > 0) ? e4.w : NEG_INF_VAL;
      *(float4*)&eBuf[r * EB_STRIDE + j0 + lane * 4] = e4;
      rmax = fmaxf(rmax, fmaxf(fmaxf(e4.x, e4.y), fmaxf(e4.z, e4.w)));
    }
    for (int m = 16; m >= 1; m >>= 1)
      rmax = fmaxf(rmax, __shfl_xor(rmax, m, 32));
    float lsum = 0.f;
    for (int j0 = 0; j0 < NN; j0 += 128) {
      float4* p = (float4*)&eBuf[r * EB_STRIDE + j0 + lane * 4];
      float4 v = *p;
      v.x = __expf(v.x - rmax);
      v.y = __expf(v.y - rmax);
      v.z = __expf(v.z - rmax);
      v.w = __expf(v.w - rmax);
      *p = v;                                       // unnormalized weights
      lsum += v.x + v.y + v.z + v.w;
    }
    for (int m = 16; m >= 1; m >>= 1) lsum += __shfl_xor(lsum, m, 32);
    if (lane == 0) rowSum[r] = lsum;
  }
  __syncthreads();

  // --- GEMM: each wave accumulates all 8 o-tiles over its 8 j-chunks ---
  const int lh  = lane >> 4;
  const int l16 = lane & 15;
  F32x8 acc[8];
  for (int t = 0; t < 8; ++t)
    for (int v = 0; v < 8; ++v) acc[t].f[v] = 0.f;

  const float* aRow = &eBuf[l16 * EB_STRIDE];   // A row = lane%16 (both halves)
  for (int c = 0; c < 8; ++c) {
    int jj = (wave * 8 + c) * 32;
    const float4* p0 = (const float4*)(aRow + jj + (lh ? 8 : 0));
    const float4* p1 = (const float4*)(aRow + jj + (lh ? 24 : 16));
    float4 f0 = p0[0], f1 = p0[1], f2 = p1[0], f3 = p1[1];
    BF16x16 af;
    af.u[0] = pack_bf16(f0.x, f0.y); af.u[1] = pack_bf16(f0.z, f0.w);
    af.u[2] = pack_bf16(f1.x, f1.y); af.u[3] = pack_bf16(f1.z, f1.w);
    af.u[4] = pack_bf16(f2.x, f2.y); af.u[5] = pack_bf16(f2.z, f2.w);
    af.u[6] = pack_bf16(f3.x, f3.y); af.u[7] = pack_bf16(f3.z, f3.w);
    for (int t = 0; t < 8; ++t) {
      int o = t * 16 + l16;
      const uint4* bp = (const uint4*)(whT + (size_t)(b * FOUT + o) * NN
                                           + jj + (lh ? 8 : 0));
      BF16x16 bf;
      bf.q[0] = bp[0];
      bf.q[1] = bp[2];
      acc[t].v = __builtin_amdgcn_wmma_f32_16x16x32_bf16(
          false, af.v, false, bf.v, (short)0, acc[t].v, false, false);
    }
  }
  __syncthreads();                 // done reading eBuf; reuse it for partials

  float* red = eBuf;               // [8 waves][16*128]
  for (int t = 0; t < 8; ++t)
    for (int v = 0; v < 8; ++v) {
      int m = lh ? v + 8 : v;
      red[wave * 2048 + m * FOUT + (t * 16 + l16)] = acc[t].f[v];
    }
  __syncthreads();

  for (int idx = tid; idx < 16 * FOUT; idx += 256) {
    int m = idx >> 7;
    float v = 0.f;
    for (int w = 0; w < 8; ++w) v += red[w * 2048 + idx];
    v /= rowSum[m];                              // softmax normalization
    v = (v > 0.f) ? v : (__expf(v) - 1.f);       // elu
    out[(size_t)(b * NN + i0 + m) * FOUT + (idx & 127)] = v;
  }
}

// ---------------------------------------------------------------------------
extern "C" void kernel_launch(void* const* d_in, const int* in_sizes, int n_in,
                              void* d_out, int out_size, void* d_ws, size_t ws_size,
                              hipStream_t stream) {
  const float* h   = (const float*)d_in[0];   // (8,2048,256)
  const int*   adj = (const int*)  d_in[1];   // (8,2048,2048)
  const float* Ww  = (const float*)d_in[2];   // (128,256)
  const float* Wb  = (const float*)d_in[3];   // (128,)
  const float* aw  = (const float*)d_in[4];   // (1,256)
  const float* ab  = (const float*)d_in[5];   // (1,)
  float* out = (float*)d_out;

  char* ws = (char*)d_ws;
  constexpr size_t WH_BYTES  = (size_t)BB * NN * FOUT * 4;   // 8,388,608
  constexpr size_t WHT_BYTES = (size_t)BB * FOUT * NN * 2;   // 4,194,304
  constexpr size_t S_BYTES   = (size_t)BB * NN * 4;          //    65,536
  float*    wh  = (float*)ws;
  uint16_t* whT = (uint16_t*)(ws + WH_BYTES);
  float*    s1  = (float*)(ws + WH_BYTES + WHT_BYTES);
  float*    s2  = (float*)(ws + WH_BYTES + WHT_BYTES + S_BYTES);
  (void)in_sizes; (void)n_in; (void)out_size; (void)ws_size;

  gat_wh  <<<dim3(NN / 64, BB), 256, 0, stream>>>(h, Ww, Wb, wh, whT);
  gat_s12 <<<dim3((BB * NN) / 256), 256, 0, stream>>>(wh, aw, s1, s2);
  gat_attn<<<dim3(NN / 16, BB), 256, 0, stream>>>(adj, s1, s2, ab, whT, out);
}
